// SMEModel_9182640079042
// MI455X (gfx1250) — matrix operand order repaired and checked by reading, not compiled, not run on hardware
//
#include <hip/hip_runtime.h>
#include <hip/hip_bf16.h>

// ---------------- problem constants ----------------
#define BB    16
#define TT    1024
#define DIN   4608
#define DRED  128
#define HH    32
#define G3    96          // 3*H
#define KCH   (DIN/32)    // 144 K-chunks of 32 for bf16 WMMA
#define TAU   12
#define QPV   1.0e4f

// ---------------- workspace layout (bytes) ----------------
// Bp  : packed bf16 B fragments  [KCH][6][32 lanes][16 bf16] = 884736 B
// bc  : composite bias f32 [96]
// GI  : input-gate preactivations f32 [B*T][96]  = 6291456 B
// HS  : GRU hidden states f32 [B*T][32]          = 2097152 B
#define BP_OFF  0
#define BC_OFF  884736
#define GI_OFF  885248
#define HS_OFF  (885248 + 6291456)

typedef __attribute__((ext_vector_type(16))) __bf16 v16bf;
typedef __attribute__((ext_vector_type(8)))  float  v8f;

__device__ __forceinline__ float sigf(float x) {
    return 1.0f / (1.0f + __expf(-x));
}
__device__ __forceinline__ float tanh_fast(float x) {
    float e = __expf(-2.0f * fabsf(x));
    float t = (1.0f - e) / (1.0f + e);
    return copysignf(t, x);
}

// ---------------------------------------------------------------------------
// Kernel 1: composite weight  w_c = w_ih @ w_dr  (96 x 4608), bf16-packed into
// WMMA B-fragment layout; composite bias bc = w_ih @ b_dr + b_ih.
// ---------------------------------------------------------------------------
__global__ __launch_bounds__(256) void prep_kernel(
    const float* __restrict__ w_dr, const float* __restrict__ b_dr,
    const float* __restrict__ w_ih, const float* __restrict__ b_ih,
    __bf16* __restrict__ Bp, float* __restrict__ bc)
{
    int idx = blockIdx.x * 256 + threadIdx.x;          // total = 96*4608
    int k = idx % DIN;
    int n = idx / DIN;

    float acc = 0.0f;
    #pragma unroll 8
    for (int r = 0; r < DRED; ++r)
        acc = fmaf(w_ih[n * DRED + r], w_dr[(size_t)r * DIN + k], acc);

    // pack into B fragment layout: B is 32x16 (KxN) per chunk,
    // lane = (n%16) + ((kk&8) ? 16 : 0), slot = kk%8 + (kk<16 ? 0 : 8)
    int kk   = k & 31;
    int kc   = k >> 5;
    int ct   = n >> 4;
    int lane = (n & 15) + ((kk & 8) ? 16 : 0);
    int slot = (kk & 7) + ((kk & 16) ? 8 : 0);
    Bp[(((size_t)kc * 6 + ct) * 32 + lane) * 16 + slot] = (__bf16)acc;

    if (idx < G3) {
        float a = 0.0f;
        #pragma unroll 8
        for (int r = 0; r < DRED; ++r)
            a = fmaf(w_ih[idx * DRED + r], b_dr[r], a);
        bc[idx] = a + b_ih[idx];
    }
}

// ---------------------------------------------------------------------------
// Kernel 2: GI[16384, 96] = bf16(x) @ bf16(w_c)^T + bc via v_wmma_f32_16x16x32_bf16.
// grid: 64 workgroups x 8 waves; each wave owns TWO 16-row tiles (32 rows) so
// every B fragment read from L2 feeds two WMMAs (halves L2 B traffic).
// ---------------------------------------------------------------------------
__global__ __launch_bounds__(256) void gemm_kernel(
    const float* __restrict__ x, const __bf16* __restrict__ Bp,
    const float* __restrict__ bc, float* __restrict__ GI)
{
    const int wave = threadIdx.x >> 5;
    const int lane = threadIdx.x & 31;
    const int rowBase = blockIdx.x * 256 + wave * 32;      // 2 row-tiles per wave
    const int arow0 = rowBase + (lane & 15);
    const int arow1 = arow0 + 16;
    const int kbase = (lane & 16) ? 8 : 0;   // A-layout: lanes 0-15 K{0-7,16-23}, 16-31 K{8-15,24-31}

    v8f acc0[6], acc1[6];
    #pragma unroll
    for (int ct = 0; ct < 6; ++ct) {
        float bias = bc[(lane & 15) + 16 * ct];
        #pragma unroll
        for (int v = 0; v < 8; ++v) { acc0[ct][v] = bias; acc1[ct][v] = bias; }
    }

    const float* ar0 = x + (size_t)arow0 * DIN;
    const float* ar1 = x + (size_t)arow1 * DIN;

    for (int kc = 0; kc < KCH; ++kc) {
        const float* p0 = ar0 + kc * 32 + kbase;
        const float* p1 = ar1 + kc * 32 + kbase;
        float4 a0 = *(const float4*)(p0);
        float4 a1 = *(const float4*)(p0 + 4);
        float4 a2 = *(const float4*)(p0 + 16);
        float4 a3 = *(const float4*)(p0 + 20);
        float4 c0 = *(const float4*)(p1);
        float4 c1 = *(const float4*)(p1 + 4);
        float4 c2 = *(const float4*)(p1 + 16);
        float4 c3 = *(const float4*)(p1 + 20);

        v16bf af0, af1;
        af0[0]  = (__bf16)a0.x; af0[1]  = (__bf16)a0.y; af0[2]  = (__bf16)a0.z; af0[3]  = (__bf16)a0.w;
        af0[4]  = (__bf16)a1.x; af0[5]  = (__bf16)a1.y; af0[6]  = (__bf16)a1.z; af0[7]  = (__bf16)a1.w;
        af0[8]  = (__bf16)a2.x; af0[9]  = (__bf16)a2.y; af0[10] = (__bf16)a2.z; af0[11] = (__bf16)a2.w;
        af0[12] = (__bf16)a3.x; af0[13] = (__bf16)a3.y; af0[14] = (__bf16)a3.z; af0[15] = (__bf16)a3.w;
        af1[0]  = (__bf16)c0.x; af1[1]  = (__bf16)c0.y; af1[2]  = (__bf16)c0.z; af1[3]  = (__bf16)c0.w;
        af1[4]  = (__bf16)c1.x; af1[5]  = (__bf16)c1.y; af1[6]  = (__bf16)c1.z; af1[7]  = (__bf16)c1.w;
        af1[8]  = (__bf16)c2.x; af1[9]  = (__bf16)c2.y; af1[10] = (__bf16)c2.z; af1[11] = (__bf16)c2.w;
        af1[12] = (__bf16)c3.x; af1[13] = (__bf16)c3.y; af1[14] = (__bf16)c3.z; af1[15] = (__bf16)c3.w;

        #pragma unroll
        for (int ct = 0; ct < 6; ++ct) {
            v16bf bfrag = *(const v16bf*)(Bp + (((size_t)kc * 6 + ct) * 32 + lane) * 16);
            acc0[ct] = __builtin_amdgcn_wmma_f32_16x16x32_bf16(
                false, af0, false, bfrag, (short)0, acc0[ct], false, false);
            acc1[ct] = __builtin_amdgcn_wmma_f32_16x16x32_bf16(
                false, af1, false, bfrag, (short)0, acc1[ct], false, false);
        }
    }

    // C/D layout: lane 0-15 -> N=lane, M=v; lane 16-31 -> N=lane-16, M=8+v
    const int mAdd = (lane >> 4) << 3;
    const int col0 = lane & 15;
    #pragma unroll
    for (int ct = 0; ct < 6; ++ct) {
        #pragma unroll
        for (int v = 0; v < 8; ++v) {
            int m = mAdd + v;
            GI[(size_t)(rowBase + m)      * G3 + col0 + 16 * ct] = acc0[ct][v];
            GI[(size_t)(rowBase + 16 + m) * G3 + col0 + 16 * ct] = acc1[ct][v];
        }
    }
}

// ---------------------------------------------------------------------------
// Kernel 3: sequential GRU, one wave32 per batch element. h[j] lives in lane j.
// w_hh rows (j, j+32, j+64) are register-resident; lane broadcasts via __shfl.
// ---------------------------------------------------------------------------
__global__ __launch_bounds__(32) void gru_kernel(
    const float* __restrict__ GI, const float* __restrict__ w_hh,
    const float* __restrict__ b_hh, float* __restrict__ HS)
{
    const int b = blockIdx.x;
    const int j = threadIdx.x;

    float wr[HH], wz[HH], wn[HH];
    #pragma unroll
    for (int k = 0; k < HH; ++k) {
        wr[k] = w_hh[(j)      * HH + k];
        wz[k] = w_hh[(j + 32) * HH + k];
        wn[k] = w_hh[(j + 64) * HH + k];
    }
    const float bhr = b_hh[j], bhz = b_hh[32 + j], bhn = b_hh[64 + j];

    float h = 0.0f;
    const float* gib = GI + (size_t)b * TT * G3;
    float*       hsb = HS + (size_t)b * TT * HH;

    for (int t = 0; t < TT; ++t) {
        const float* g = gib + (size_t)t * G3;
        float gi_r = g[j], gi_z = g[32 + j], gi_n = g[64 + j];

        float ar = 0.0f, az = 0.0f, an = 0.0f;
        #pragma unroll
        for (int k = 0; k < HH; ++k) {
            float hk = __shfl(h, k, 32);
            ar = fmaf(wr[k], hk, ar);
            az = fmaf(wz[k], hk, az);
            an = fmaf(wn[k], hk, an);
        }
        float r  = sigf(gi_r + ar + bhr);
        float z  = sigf(gi_z + az + bhz);
        float n  = tanh_fast(gi_n + r * (an + bhn));
        h = fmaf(z, h - n, n);            // (1-z)*n + z*h
        hsb[(size_t)t * HH + j] = h;
    }
}

// ---------------------------------------------------------------------------
// Kernel 4: q head + windowed min / exp-weighted pooling + masked mean + MLP tail.
// One 256-thread block per batch element; q[] lives in LDS.
// ---------------------------------------------------------------------------
__global__ __launch_bounds__(256) void final_kernel(
    const float* __restrict__ HS, const int* __restrict__ x_len,
    const float* __restrict__ w_reg, const float* __restrict__ b_reg,
    const float* __restrict__ w_nlm1, const float* __restrict__ b_nlm1,
    const float* __restrict__ w_nlm2, const float* __restrict__ b_nlm2,
    const float* __restrict__ w_lm, const float* __restrict__ b_lm,
    float* __restrict__ out)
{
    __shared__ float q[TT];
    __shared__ float red[256];

    const int b   = blockIdx.x;
    const int tid = threadIdx.x;
    const int len = x_len[b];

    float wv[HH];
    #pragma unroll
    for (int k = 0; k < HH; ++k) wv[k] = w_reg[k];
    const float br = b_reg[0];

    #pragma unroll
    for (int i = 0; i < TT / 256; ++i) {
        int t = tid + i * 256;
        const float* hp = HS + ((size_t)b * TT + t) * HH;
        float s = br;
        #pragma unroll
        for (int k = 0; k < HH; ++k) s = fmaf(hp[k], wv[k], s);
        q[t] = s;
    }
    __syncthreads();

    float local = 0.0f;
    #pragma unroll
    for (int i = 0; i < TT / 256; ++i) {
        int t = tid + i * 256;
        // l[t] = min(q[max(0,t-tau+1) .. t])
        float l = q[t];
        #pragma unroll
        for (int k = 1; k < TAU; ++k) {
            int idx = t - k;
            if (idx >= 0) l = fminf(l, q[idx]);
        }
        // exp-weighted forward average over q_eff[t .. t+tau-1], pad QP
        float num = 0.0f, den = 0.0f;
        #pragma unroll
        for (int k = 0; k < TAU; ++k) {
            int idx = t + k;
            float w = (idx < TT && idx < len) ? q[idx] : QPV;
            float e = __expf(-w);          // exp(-1e4) underflows to 0 like ref
            num = fmaf(w, e, num);
            den += e;
        }
        bool valid = t < len;
        float m  = valid ? (num / den) : 0.0f;   // mean/mean ratio == sum/sum
        float qh = 0.5f * m + 0.5f * l;          // BETA = 0.5
        if (valid) local += qh;
    }

    red[tid] = local;
    __syncthreads();
    for (int s = 128; s > 0; s >>= 1) {
        if (tid < s) red[tid] += red[tid + s];
        __syncthreads();
    }

    if (tid == 0) {
        float rel      = red[0] / (float)len;
        float relative = sigf(rel);
        float mapped   = sigf(relative * w_nlm1[0] + b_nlm1[0]) * w_nlm2[0] + b_nlm2[0];
        float aligned  = mapped * w_lm[0] + b_lm[0];
        out[b]          = relative;
        out[BB + b]     = mapped;
        out[2 * BB + b] = aligned;
    }
}

// ---------------------------------------------------------------------------
extern "C" void kernel_launch(void* const* d_in, const int* in_sizes, int n_in,
                              void* d_out, int out_size, void* d_ws, size_t ws_size,
                              hipStream_t stream) {
    const float* x      = (const float*)d_in[0];
    const int*   x_len  = (const int*)  d_in[1];
    const float* w_dr   = (const float*)d_in[2];
    const float* b_dr   = (const float*)d_in[3];
    const float* w_ih   = (const float*)d_in[4];
    const float* w_hh   = (const float*)d_in[5];
    const float* b_ih   = (const float*)d_in[6];
    const float* b_hh   = (const float*)d_in[7];
    const float* w_reg  = (const float*)d_in[8];
    const float* b_reg  = (const float*)d_in[9];
    const float* w_nlm1 = (const float*)d_in[10];
    const float* b_nlm1 = (const float*)d_in[11];
    const float* w_nlm2 = (const float*)d_in[12];
    const float* b_nlm2 = (const float*)d_in[13];
    const float* w_lm   = (const float*)d_in[14];
    const float* b_lm   = (const float*)d_in[15];

    char* ws = (char*)d_ws;
    __bf16* Bp = (__bf16*)(ws + BP_OFF);
    float*  bc = (float*) (ws + BC_OFF);
    float*  GI = (float*) (ws + GI_OFF);
    float*  HS = (float*) (ws + HS_OFF);
    float*  out = (float*)d_out;

    prep_kernel <<<(G3 * DIN) / 256, 256, 0, stream>>>(w_dr, b_dr, w_ih, b_ih, Bp, bc);
    gemm_kernel <<<(BB * TT) / 256, 256, 0, stream>>>(x, Bp, bc, GI);
    gru_kernel  <<<BB, 32, 0, stream>>>(GI, w_hh, b_hh, HS);
    final_kernel<<<BB, 256, 0, stream>>>(HS, x_len, w_reg, b_reg,
                                         w_nlm1, b_nlm1, w_nlm2, b_nlm2,
                                         w_lm, b_lm, out);
}
